// PerformerAttention_73151882986027
// MI455X (gfx1250) — matrix operand order repaired and checked by reading, not compiled, not run on hardware
//
#include <hip/hip_runtime.h>
#include <hip/hip_bf16.h>
#include <float.h>

typedef __bf16 bf16_t;
typedef __attribute__((ext_vector_type(16))) __bf16 v16bf;
typedef __attribute__((ext_vector_type(8)))  float  v8f;

#define B_   8
#define T_   2048
#define F_   512
#define H_   8
#define DK_  64
#define M_   256
#define BH_  (B_ * H_)     // 64
#define BT_  (B_ * T_)     // 16384
#define RATIO_ 0.0625f     // M^-0.5
#define EPS_   1e-4f
#define GSCALE_ 0.125f     // DK^-0.5 (folds scale*dn from reference)

__device__ __forceinline__ int lane_id() { return (int)(threadIdx.x & 31); }

// ---- WMMA fragment loaders (CDNA5 bf16 16x16x32 layouts) ----
// A (16xK tile, row-major, ld elements): lane l -> row l&15,
//   elems 0..7  = A[row, kb+0..7],  elems 8..15 = A[row, kb+16..23], kb=(l>>4)*8
__device__ __forceinline__ v16bf load_a_bf16(const bf16_t* base, int ld) {
  const int l = lane_id();
  const bf16_t* p = base + (size_t)(l & 15) * ld + ((l >> 4) << 3);
  v16bf f;
#pragma unroll
  for (int i = 0; i < 8; ++i) { f[i] = p[i]; f[8 + i] = p[16 + i]; }
  return f;
}

__device__ __forceinline__ v16bf load_a_f32(const float* base, int ld) {
  const int l = lane_id();
  const float* p = base + (size_t)(l & 15) * ld + ((l >> 4) << 3);
  v16bf f;
#pragma unroll
  for (int i = 0; i < 8; ++i) { f[i] = (bf16_t)p[i]; f[8 + i] = (bf16_t)p[16 + i]; }
  return f;
}

// B (Kx16 tile) sourced from [N][K] storage: lane l -> col n=l&15,
//   elems 0..7 = Nmat[n, kb+0..7], elems 8..15 = Nmat[n, kb+16..23], kb=(l>>4)*8
__device__ __forceinline__ v16bf load_b_bf16(const bf16_t* base, int ld) {
  const int l = lane_id();
  const bf16_t* p = base + (size_t)(l & 15) * ld + ((l >> 4) << 3);
  v16bf f;
#pragma unroll
  for (int i = 0; i < 8; ++i) { f[i] = p[i]; f[8 + i] = p[16 + i]; }
  return f;
}

__device__ __forceinline__ v8f wmma_bf16(v16bf a, v16bf b, v8f c) {
  return __builtin_amdgcn_wmma_f32_16x16x32_bf16(false, a, false, b, (short)0, c,
                                                 false, false);
}

// ---------------- kernel 1: weight / proj conversion + stab init -------------
__global__ void cvt_weights_kernel(const float* Wq, const float* Wk,
                                   const float* Wv, const float* Wo,
                                   const float* proj,
                                   bf16_t* wq, bf16_t* wk, bf16_t* wv,
                                   bf16_t* wo, bf16_t* pj, float* stab) {
  int i = blockIdx.x * blockDim.x + threadIdx.x;
  if (i < F_ * F_) {
    wq[i] = (bf16_t)Wq[i];
    wk[i] = (bf16_t)Wk[i];
    wv[i] = (bf16_t)Wv[i];
    wo[i] = (bf16_t)Wo[i];
  }
  if (i < M_ * DK_) pj[i] = (bf16_t)proj[i];
  if (i == 0) *stab = -FLT_MAX;
}

// ---------------- shared epilogue: one 16x16 fp32 tile -----------------------
// MODE 0: *scale, bf16 head-major  [bh, T, DK]   (Q / K feature input g)
// MODE 1: bf16 transposed          [bh, DK, T]   (V)
// MODE 2: fp32 row-major           [BT, F]       (final output)
template <int MODE>
__device__ __forceinline__ void store_proj_tile(const v8f& acc, int rowt,
                                                int colt, const float* bias,
                                                bf16_t* outB, float* outF,
                                                float scale) {
  const int l = lane_id();
  const int col = colt + (l & 15);
  const int rb = (l >> 4) * 8;
  const float bv = bias[col];
#pragma unroll
  for (int r = 0; r < 8; ++r) {
    const int n = rowt + rb + r;
    const float v = (acc[r] + bv) * scale;
    const int b_ = n >> 11, t = n & (T_ - 1);
    const int h = col >> 6, d = col & 63;
    if (MODE == 0)
      outB[((size_t)(b_ * H_ + h) * T_ + t) * DK_ + d] = (bf16_t)v;
    else if (MODE == 1)
      outB[((size_t)(b_ * H_ + h) * DK_ + d) * T_ + t] = (bf16_t)v;
    else
      outF[(size_t)n * F_ + col] = v;
  }
}

// ---------------- kernel 2: GEMM  D[n,f] = X[n,:].W[f,:] + bias[f] -----------
// block = 128 (4 waves, 2x2), block tile 64x64, wave tile 32x32 (2x2 WMMA).
template <int MODE, bool AF32>
__global__ void proj_gemm_kernel(const float* Xf, const bf16_t* Xb,
                                 const bf16_t* W, const float* bias,
                                 bf16_t* outB, float* outF, float scale) {
  const int w = (int)(threadIdx.x >> 5);
  const int row0 = blockIdx.x * 64 + (w >> 1) * 32;
  const int col0 = blockIdx.y * 64 + (w & 1) * 32;
  v8f a00 = {}, a01 = {}, a10 = {}, a11 = {};
#pragma unroll 4
  for (int k = 0; k < F_; k += 32) {
    v16bf fa0, fa1;
    if (AF32) {
      fa0 = load_a_f32(Xf + (size_t)row0 * F_ + k, F_);
      fa1 = load_a_f32(Xf + (size_t)(row0 + 16) * F_ + k, F_);
    } else {
      fa0 = load_a_bf16(Xb + (size_t)row0 * F_ + k, F_);
      fa1 = load_a_bf16(Xb + (size_t)(row0 + 16) * F_ + k, F_);
    }
    const v16bf fb0 = load_b_bf16(W + (size_t)col0 * F_ + k, F_);
    const v16bf fb1 = load_b_bf16(W + (size_t)(col0 + 16) * F_ + k, F_);
    a00 = wmma_bf16(fa0, fb0, a00);
    a01 = wmma_bf16(fa0, fb1, a01);
    a10 = wmma_bf16(fa1, fb0, a10);
    a11 = wmma_bf16(fa1, fb1, a11);
  }
  store_proj_tile<MODE>(a00, row0, col0, bias, outB, outF, scale);
  store_proj_tile<MODE>(a01, row0, col0 + 16, bias, outB, outF, scale);
  store_proj_tile<MODE>(a10, row0 + 16, col0, bias, outB, outF, scale);
  store_proj_tile<MODE>(a11, row0 + 16, col0 + 16, bias, outB, outF, scale);
}

// ---------------- kernel 3: feature map  dash = g @ proj^T -------------------
// mode 0: k pass 1 -> global atomic max of dash into *stab
// mode 1: q        -> qp[bh,T,M]  = ratio*(exp(dash-diag-rowmax)+eps)
// mode 2: k pass 2 -> kpT[bh,M,T] = ratio*(exp(dash-diag-*stab)+eps)
__global__ void features_kernel(const bf16_t* g, const bf16_t* pj,
                                bf16_t* qp, bf16_t* kpT, float* stab,
                                int mode) {
  __shared__ float dash[16][M_];
  __shared__ float rowred[16];
  const int bh = blockIdx.y;
  const int row0 = blockIdx.x * 16;
  const bf16_t* gb = g + ((size_t)bh * T_ + row0) * DK_;
  const v16bf a0 = load_a_bf16(gb, DK_);
  const v16bf a1 = load_a_bf16(gb + 32, DK_);
  const int l = lane_id();
#pragma unroll
  for (int mt = 0; mt < 16; ++mt) {
    v8f acc = {};
    v16bf b0 = load_b_bf16(pj + (size_t)mt * 16 * DK_, DK_);
    v16bf b1 = load_b_bf16(pj + (size_t)mt * 16 * DK_ + 32, DK_);
    acc = wmma_bf16(a0, b0, acc);
    acc = wmma_bf16(a1, b1, acc);
    const int col = mt * 16 + (l & 15);
    const int rb = (l >> 4) * 8;
#pragma unroll
    for (int r = 0; r < 8; ++r) dash[rb + r][col] = acc[r];
  }
  __syncthreads();
  if (l < 16) {
    const int row = l;
    float ss = 0.f;
    const bf16_t* gr = gb + (size_t)row * DK_;
#pragma unroll
    for (int d = 0; d < DK_; ++d) { float x = (float)gr[d]; ss += x * x; }
    const float diag = 0.5f * ss;
    if (mode == 0) {
      float mx = -FLT_MAX;
      for (int m = 0; m < M_; ++m) mx = fmaxf(mx, dash[row][m]);
      rowred[row] = mx;
    } else if (mode == 1) {
      float mx = -FLT_MAX;
      for (int m = 0; m < M_; ++m) mx = fmaxf(mx, dash[row][m]);
      const size_t o = ((size_t)bh * T_ + row0 + row) * M_;
      for (int m = 0; m < M_; ++m)
        qp[o + m] = (bf16_t)(RATIO_ * (__expf(dash[row][m] - diag - mx) + EPS_));
    } else {
      const float s = *stab;
      const size_t o = (size_t)bh * M_ * T_ + (size_t)(row0 + row);
      for (int m = 0; m < M_; ++m)
        kpT[o + (size_t)m * T_] =
            (bf16_t)(RATIO_ * (__expf(dash[row][m] - diag - s) + EPS_));
    }
  }
  if (mode == 0) {
    __syncthreads();
    if (threadIdx.x == 0) {
      float mx = rowred[0];
#pragma unroll
      for (int i = 1; i < 16; ++i) mx = fmaxf(mx, rowred[i]);
      if (mx >= 0.f) atomicMax((int*)stab, __float_as_int(mx));
      else           atomicMin((unsigned int*)stab, __float_as_uint(mx));
    }
  }
}

// ---------------- kernel 4: ctxT[d,m] = sum_n vT[d,n]*kpT[m,n] ---------------
// block 128 (4 waves 2x2), block tile 64(d) x 64(m), wave tile 32x32.
__global__ void context_kernel(const bf16_t* vT, const bf16_t* kpT,
                               bf16_t* ctxT) {
  const int bh = blockIdx.y;
  const int w = (int)(threadIdx.x >> 5);
  const int d0 = (w >> 1) * 32;
  const int m0 = blockIdx.x * 64 + (w & 1) * 32;
  const bf16_t* vb = vT + ((size_t)bh * DK_ + d0) * T_;
  const bf16_t* kb = kpT + ((size_t)bh * M_ + m0) * T_;
  v8f a00 = {}, a01 = {}, a10 = {}, a11 = {};
  for (int n = 0; n < T_; n += 32) {
    __builtin_prefetch(vb + n + 256, 0, 1);
    __builtin_prefetch(kb + n + 256, 0, 1);
    const v16bf fa0 = load_a_bf16(vb + n, T_);
    const v16bf fa1 = load_a_bf16(vb + (size_t)16 * T_ + n, T_);
    const v16bf fb0 = load_b_bf16(kb + n, T_);
    const v16bf fb1 = load_b_bf16(kb + (size_t)16 * T_ + n, T_);
    a00 = wmma_bf16(fa0, fb0, a00);
    a01 = wmma_bf16(fa0, fb1, a01);
    a10 = wmma_bf16(fa1, fb0, a10);
    a11 = wmma_bf16(fa1, fb1, a11);
  }
  const int l = lane_id();
  const int rb = (l >> 4) * 8;
#pragma unroll
  for (int r = 0; r < 8; ++r) {
    const size_t r0 = (size_t)bh * DK_ + d0 + rb + r;
    const size_t r1 = r0 + 16;
    ctxT[r0 * M_ + m0 + (l & 15)]      = (bf16_t)a00[r];
    ctxT[r0 * M_ + m0 + 16 + (l & 15)] = (bf16_t)a01[r];
    ctxT[r1 * M_ + m0 + (l & 15)]      = (bf16_t)a10[r];
    ctxT[r1 * M_ + m0 + 16 + (l & 15)] = (bf16_t)a11[r];
  }
}

// ---------------- kernel 5: k_sum[bh,m] = sum_t kpT[bh,m,t] ------------------
__global__ void ksum_kernel(const bf16_t* kpT, float* ksum) {
  __shared__ float red[128];
  const int bh = blockIdx.y, m = blockIdx.x;
  const bf16_t* p = kpT + ((size_t)bh * M_ + m) * T_;
  float s = 0.f;
  for (int t = threadIdx.x; t < T_; t += 128) s += (float)p[t];
  red[threadIdx.x] = s;
  __syncthreads();
  for (int w = 64; w > 0; w >>= 1) {
    if ((int)threadIdx.x < w) red[threadIdx.x] += red[threadIdx.x + w];
    __syncthreads();
  }
  if (threadIdx.x == 0) ksum[bh * M_ + m] = red[0];
}

// ---------------- kernel 6: d_inv[n] = 1 / (qp[n,:] . k_sum) -----------------
__global__ void dinv_kernel(const bf16_t* qp, const float* ksum, float* dinv) {
  const int idx = blockIdx.x * blockDim.x + threadIdx.x;  // over BH*T
  const int bh = idx >> 11;
  const bf16_t* q = qp + (size_t)idx * M_;
  const float* ks = ksum + bh * M_;
  float s = 0.f;
#pragma unroll 8
  for (int m = 0; m < M_; ++m) s += (float)q[m] * ks[m];
  dinv[idx] = 1.0f / s;
}

// ---------------- kernel 7: O[n,d] = dinv[n]*sum_m qp[n,m]*ctxT[d,m] ---------
// block 128 (4 waves 2x2), block tile 64(n) x 64(d), wave tile 32x32.
__global__ void outcore_kernel(const bf16_t* qp, const bf16_t* ctxT,
                               const float* dinv, bf16_t* O) {
  const int bh = blockIdx.y;
  const int w = (int)(threadIdx.x >> 5);
  const int row0 = blockIdx.x * 64 + (w >> 1) * 32;
  const int d0 = (w & 1) * 32;
  const bf16_t* ab = qp + ((size_t)bh * T_ + row0) * M_;
  const bf16_t* bb = ctxT + ((size_t)bh * DK_ + d0) * M_;
  v8f a00 = {}, a01 = {}, a10 = {}, a11 = {};
#pragma unroll
  for (int k = 0; k < M_; k += 32) {
    const v16bf fa0 = load_a_bf16(ab + k, M_);
    const v16bf fa1 = load_a_bf16(ab + (size_t)16 * M_ + k, M_);
    const v16bf fb0 = load_b_bf16(bb + k, M_);
    const v16bf fb1 = load_b_bf16(bb + (size_t)16 * M_ + k, M_);
    a00 = wmma_bf16(fa0, fb0, a00);
    a01 = wmma_bf16(fa0, fb1, a01);
    a10 = wmma_bf16(fa1, fb0, a10);
    a11 = wmma_bf16(fa1, fb1, a11);
  }
  const int l = lane_id();
  const int rb = (l >> 4) * 8;
  const int b_ = bh >> 3, h = bh & 7;
#pragma unroll
  for (int r = 0; r < 8; ++r) {
    const int t0 = row0 + rb + r;
    const int t1 = t0 + 16;
    const float di0 = dinv[bh * T_ + t0];
    const float di1 = dinv[bh * T_ + t1];
    const size_t o0 = (size_t)(b_ * T_ + t0) * F_ + h * DK_;
    const size_t o1 = (size_t)(b_ * T_ + t1) * F_ + h * DK_;
    O[o0 + d0 + (l & 15)]      = (bf16_t)(a00[r] * di0);
    O[o0 + d0 + 16 + (l & 15)] = (bf16_t)(a01[r] * di0);
    O[o1 + d0 + (l & 15)]      = (bf16_t)(a10[r] * di1);
    O[o1 + d0 + 16 + (l & 15)] = (bf16_t)(a11[r] * di1);
  }
}

// ---------------------------------------------------------------------------
extern "C" void kernel_launch(void* const* d_in, const int* in_sizes, int n_in,
                              void* d_out, int out_size, void* d_ws,
                              size_t ws_size, hipStream_t stream) {
  const float* query = (const float*)d_in[0];
  const float* key_  = (const float*)d_in[1];
  const float* value = (const float*)d_in[2];
  /* d_in[3] = mask: unused (matches reference) */
  const float* Wq = (const float*)d_in[4];
  const float* bq = (const float*)d_in[5];
  const float* Wk = (const float*)d_in[6];
  const float* bk = (const float*)d_in[7];
  const float* Wv = (const float*)d_in[8];
  const float* bv = (const float*)d_in[9];
  const float* Wo = (const float*)d_in[10];
  const float* bo = (const float*)d_in[11];
  const float* proj = (const float*)d_in[12];

  char* ws = (char*)d_ws;
  size_t off = 0;
  auto take = [&](size_t bytes) -> char* {
    char* p = ws + off;
    off += (bytes + 255) & ~(size_t)255;
    return p;
  };
  float*  stab = (float*)take(4);
  bf16_t* wq   = (bf16_t*)take((size_t)F_ * F_ * 2);
  bf16_t* wk   = (bf16_t*)take((size_t)F_ * F_ * 2);
  bf16_t* wv   = (bf16_t*)take((size_t)F_ * F_ * 2);
  bf16_t* wo   = (bf16_t*)take((size_t)F_ * F_ * 2);
  bf16_t* pj   = (bf16_t*)take((size_t)M_ * DK_ * 2);
  bf16_t* qg   = (bf16_t*)take((size_t)BH_ * T_ * DK_ * 2);
  bf16_t* kg   = (bf16_t*)take((size_t)BH_ * T_ * DK_ * 2);
  bf16_t* vT   = (bf16_t*)take((size_t)BH_ * T_ * DK_ * 2);
  bf16_t* qp   = (bf16_t*)take((size_t)BH_ * T_ * M_ * 2);
  bf16_t* kpT  = (bf16_t*)take((size_t)BH_ * T_ * M_ * 2);
  bf16_t* ctxT = (bf16_t*)take((size_t)BH_ * DK_ * M_ * 2);
  float*  ksum = (float*)take((size_t)BH_ * M_ * 4);
  float*  dinv = (float*)take((size_t)BH_ * T_ * 4);
  bf16_t* Obuf = (bf16_t*)take((size_t)BT_ * F_ * 2);
  (void)ws_size; (void)in_sizes; (void)n_in; (void)out_size;

  // 1. weights -> bf16, stab init
  cvt_weights_kernel<<<(F_ * F_ + 255) / 256, 256, 0, stream>>>(
      Wq, Wk, Wv, Wo, proj, wq, wk, wv, wo, pj, stab);

  // 2. projections (block tile 64x64)
  const dim3 gp(BT_ / 64, F_ / 64);
  proj_gemm_kernel<0, true><<<gp, 128, 0, stream>>>(query, nullptr, wq, bq, qg,
                                                    nullptr, GSCALE_);
  proj_gemm_kernel<0, true><<<gp, 128, 0, stream>>>(key_, nullptr, wk, bk, kg,
                                                    nullptr, GSCALE_);
  proj_gemm_kernel<1, true><<<gp, 128, 0, stream>>>(value, nullptr, wv, bv, vT,
                                                    nullptr, 1.0f);

  // 3. feature maps
  const dim3 gf(T_ / 16, BH_);
  features_kernel<<<gf, 32, 0, stream>>>(kg, pj, nullptr, nullptr, stab, 0);
  features_kernel<<<gf, 32, 0, stream>>>(kg, pj, nullptr, kpT, stab, 2);
  features_kernel<<<gf, 32, 0, stream>>>(qg, pj, qp, nullptr, stab, 1);

  // 4. linear attention core
  ksum_kernel<<<dim3(M_, BH_), 128, 0, stream>>>(kpT, ksum);
  context_kernel<<<dim3(M_ / 64, BH_), 128, 0, stream>>>(vT, kpT, ctxT);
  dinv_kernel<<<(BH_ * T_) / 256, 256, 0, stream>>>(qp, ksum, dinv);
  outcore_kernel<<<dim3(T_ / 64, BH_), 128, 0, stream>>>(qp, ctxT, dinv, Obuf);

  // 5. output projection -> fp32 d_out
  proj_gemm_kernel<2, false><<<gp, 128, 0, stream>>>(nullptr, Obuf, wo, bo,
                                                     nullptr, (float*)d_out,
                                                     1.0f);
}
